// SimpleTPTransformer_78632261255812
// MI455X (gfx1250) — matrix operand order repaired and checked by reading, not compiled, not run on hardware
//
#include <hip/hip_runtime.h>
#include <hip/hip_bf16.h>
#include <math.h>

// ---------------------------------------------------------------------------
// SimpleTPTransformer forward for MI455X (gfx1250): bf16 WMMA everywhere,
// Tensor Data Mover (TDM) staging of GEMM tiles into LDS when available.
// ---------------------------------------------------------------------------

typedef __bf16 bf16_t;
typedef __attribute__((ext_vector_type(8)))  bf16_t bfx8;
typedef __attribute__((ext_vector_type(16))) bf16_t bfx16;
typedef __attribute__((ext_vector_type(8)))  float  fx8;
typedef __attribute__((ext_vector_type(4)))  unsigned int v4u;
typedef __attribute__((ext_vector_type(4)))  int v4i_t;
typedef __attribute__((ext_vector_type(8)))  int v8i_t;

#define D_MODEL 2048
#define N_HEADS 16
#define HEAD_DIM 128
#define D_FF    8192
#define VOCAB   32000
#define SEQ     2048
#define BATCH   2
#define NTOK    (BATCH*SEQ)
#define EPSRN   1e-6f

// --- TDM availability / arity probe -----------------------------------------
#if defined(__has_builtin)
#  if __has_builtin(__builtin_amdgcn_tensor_load_to_lds)
#    define GEMM_USE_TDM 1
#  endif
#endif
#if __has_include(<hip/amd_detail/amd_gfx1250_TDM.h>)
#  define TDM_SIX_ARGS 1   // therock-10.0 headers => clang-23 6-arg builtin
#endif

static __device__ __forceinline__ bf16_t to_bf(float f) { return (bf16_t)f; }

union ABfrag { bfx16 v; bfx8 h[2]; };

static __device__ __forceinline__ fx8 wmma_bf16(bfx16 a, bfx16 b, fx8 c) {
    // D = A(16x32 bf16) x B(32x16 bf16) + C(16x16 f32)
    return __builtin_amdgcn_wmma_f32_16x16x32_bf16(false, a, false, b,
                                                   (short)0, c, false, false);
}

#if defined(GEMM_USE_TDM)
static __device__ __forceinline__ void tdm_load(v4u g0, v8i_t g1) {
    v4i_t z4 = {0, 0, 0, 0};
#if defined(TDM_SIX_ARGS)
    v8i_t z8 = {0, 0, 0, 0, 0, 0, 0, 0};
    __builtin_amdgcn_tensor_load_to_lds(g0, g1, z4, z4, z8, 0);
#else
    __builtin_amdgcn_tensor_load_to_lds(g0, g1, z4, z4, 0);
#endif
}
#endif

// ---------------------------------------------------------------------------
// fp32 -> bf16 conversion (weights), 4 elems/thread
// ---------------------------------------------------------------------------
__global__ __launch_bounds__(256)
void cvt_f32_to_bf16(const float* __restrict__ in, bf16_t* __restrict__ out, int n) {
    int i = (blockIdx.x * 256 + threadIdx.x) * 4;
    if (i + 3 < n) {
        float4 v = *(const float4*)(in + i);
        out[i + 0] = to_bf(v.x);
        out[i + 1] = to_bf(v.y);
        out[i + 2] = to_bf(v.z);
        out[i + 3] = to_bf(v.w);
    }
}

// ---------------------------------------------------------------------------
// Embedding gather: x[n, :] = tok_emb[tokens[n], :]
// ---------------------------------------------------------------------------
__global__ __launch_bounds__(256)
void embed_kernel(const int* __restrict__ tok, const float* __restrict__ emb,
                  float* __restrict__ x) {
    const int n = blockIdx.x;
    const int t = tok[n];
    const float4* src = (const float4*)(emb + (size_t)t * D_MODEL);
    float4* dst = (float4*)(x + (size_t)n * D_MODEL);
    #pragma unroll 2
    for (int i = threadIdx.x; i < D_MODEL / 4; i += 256) dst[i] = src[i];
}

// ---------------------------------------------------------------------------
// RMSNorm: out_bf16 = w * x * rsqrt(mean(x^2)+eps), one block per row
// ---------------------------------------------------------------------------
__global__ __launch_bounds__(256)
void rmsnorm_kernel(const float* __restrict__ X, const float* __restrict__ W,
                    bf16_t* __restrict__ out) {
    const int row = blockIdx.x;
    const float4* x4 = (const float4*)(X + (size_t)row * D_MODEL);
    float4 a = x4[threadIdx.x];
    float4 b = x4[threadIdx.x + 256];
    float ss = a.x*a.x + a.y*a.y + a.z*a.z + a.w*a.w
             + b.x*b.x + b.y*b.y + b.z*b.z + b.w*b.w;
    #pragma unroll
    for (int m = 16; m >= 1; m >>= 1) ss += __shfl_xor(ss, m, 32);
    __shared__ float red[8];
    if ((threadIdx.x & 31) == 0) red[threadIdx.x >> 5] = ss;
    __syncthreads();
    float tot = red[0]+red[1]+red[2]+red[3]+red[4]+red[5]+red[6]+red[7];
    float sc = rsqrtf(tot * (1.0f / D_MODEL) + EPSRN);
    const float4* w4 = (const float4*)W;
    float4 wa = w4[threadIdx.x], wb = w4[threadIdx.x + 256];
    bf16_t* o = out + (size_t)row * D_MODEL;
    int i0 = threadIdx.x * 4, i1 = (threadIdx.x + 256) * 4;
    o[i0+0] = to_bf(a.x * wa.x * sc); o[i0+1] = to_bf(a.y * wa.y * sc);
    o[i0+2] = to_bf(a.z * wa.z * sc); o[i0+3] = to_bf(a.w * wa.w * sc);
    o[i1+0] = to_bf(b.x * wb.x * sc); o[i1+1] = to_bf(b.y * wb.y * sc);
    o[i1+2] = to_bf(b.z * wb.z * sc); o[i1+3] = to_bf(b.w * wb.w * sc);
}

// ---------------------------------------------------------------------------
// WMMA GEMM: C[M,N] = A[M,K] * B[N,K]^T    (A,B bf16, K-contiguous rows)
// Tiles: 128x128x32, 256 threads = 8 waves (2 along M x 4 along N),
// each wave owns a 64x32 sub-tile = 4x2 WMMA C-fragments.
// Tile staging:
//   TDM path: wave 0 issues tensor_load_to_lds for the NEXT stage's A/B
//     tiles (128x32 bf16, DMA writes the bank-skew padding directly via
//     pad_interval=3 (16 DWORDs data) / pad_amount=3 (4 DWORDs pad) = our
//     40-elem LDS row stride), then s_wait_tensorcnt 0 + one barrier/stage.
//   Fallback: register-staged global loads + ds stores, double barrier.
// ---------------------------------------------------------------------------
#define GBM 128
#define GBN 128
#define GBK 32
#define GLD 40   // LDS row stride (elems): 80B = 64B data + 16B pad

enum { EPI_BF16 = 0, EPI_F32 = 1, EPI_RES = 2, EPI_SILU = 3 };

template <int EPI>
__global__ __launch_bounds__(256)
void gemm_bf16_wmma(const bf16_t* __restrict__ A, const bf16_t* __restrict__ B,
                    void* __restrict__ Cp, int M, int N, int K) {
    __shared__ bf16_t As[2][GBM * GLD];
    __shared__ bf16_t Bs[2][GBN * GLD];

    const int tid  = threadIdx.x;
    const int wave = tid >> 5, lane = tid & 31;
    const int lh   = lane >> 4, l16 = lane & 15;     // C/A/B layout half + sublane
    const int wm   = wave >> 2, wn = wave & 3;
    const int bm   = blockIdx.y * GBM, bnb = blockIdx.x * GBN;

    fx8 acc[4][2] = {};
    const int nst = K / GBK;

#if defined(GEMM_USE_TDM)
    // ---- Tensor Data Mover staging --------------------------------------
    const unsigned ldsA[2] = { (unsigned)(size_t)(&As[0][0]),
                               (unsigned)(size_t)(&As[1][0]) };
    const unsigned ldsB[2] = { (unsigned)(size_t)(&Bs[0][0]),
                               (unsigned)(size_t)(&Bs[1][0]) };
    // D# group 1 (shared by A and B tiles): data_size=2B, pad 16B per 64B row,
    // tensor_dim0=K, tensor_dim1 large, tile 32x128, dim0 stride = K.
    v8i_t g1;
    g1[0] = (1 << 16)            // data_size = 1 -> 2 bytes
          | (1 << 20)            // pad_enable
          | (3 << 22)            // pad_interval: 16 DWORDs of data
          | (3 << 25);           // pad_amount:   4 DWORDs of pad
    g1[1] = (K & 0xFFFF) << 16;                 // tensor_dim0 low16
    g1[2] = ((unsigned)K >> 16) | (1u << 20);   // tensor_dim0 hi / tensor_dim1 lo
    g1[3] = (GBK << 16);                        // tensor_dim1 hi=0 | tile_dim0=32
    g1[4] = GBM;                                // tile_dim1 = 128, tile_dim2 = 0
    g1[5] = K;                                  // tensor_dim0_stride low32
    g1[6] = 0;
    g1[7] = 0;

    const bf16_t* abase = A + (size_t)bm  * K;
    const bf16_t* bbase = B + (size_t)bnb * K;
    const bool issuer = (tid < 32);             // wave 0 drives the TDM

    auto tdm_issue = [&](const bf16_t* gptr, unsigned lds) {
        const unsigned long long ga = (unsigned long long)(size_t)gptr;
        v4u g0;
        g0[0] = 1u;                                           // count=1, user
        g0[1] = lds;                                          // lds_addr
        g0[2] = (unsigned)ga;                                 // global[31:0]
        g0[3] = (unsigned)((ga >> 32) & 0x01FFFFFFu)          // global[56:32]
              | (2u << 30);                                   // type=2 (image)
        tdm_load(g0, g1);
    };

    if (issuer) {
        tdm_issue(abase, ldsA[0]);
        tdm_issue(bbase, ldsB[0]);
        __builtin_amdgcn_s_wait_tensorcnt(0);
    }
    __syncthreads();
#else
    // ---- fallback: register-staged global loads -------------------------
    const int lrow = tid >> 1;
    const int lcol = (tid & 1) * 16;
    const bf16_t* ag = A + (size_t)(bm  + lrow) * K + lcol;
    const bf16_t* bg = B + (size_t)(bnb + lrow) * K + lcol;
    bfx8 ra0, ra1, rb0, rb1;
    ra0 = *(const bfx8*)ag;       ra1 = *(const bfx8*)(ag + 8);
    rb0 = *(const bfx8*)bg;       rb1 = *(const bfx8*)(bg + 8);
    {
        bf16_t* da = &As[0][lrow * GLD + lcol];
        bf16_t* db = &Bs[0][lrow * GLD + lcol];
        *(bfx8*)da = ra0; *(bfx8*)(da + 8) = ra1;
        *(bfx8*)db = rb0; *(bfx8*)(db + 8) = rb1;
    }
    __syncthreads();
#endif

    for (int s = 0; s < nst; ++s) {
        const int cur = s & 1;

#if defined(GEMM_USE_TDM)
        if (s + 1 < nst && issuer) {    // DMA next stage while we compute
            tdm_issue(abase + (size_t)(s + 1) * GBK, ldsA[cur ^ 1]);
            tdm_issue(bbase + (size_t)(s + 1) * GBK, ldsB[cur ^ 1]);
        }
#else
        if (s + 1 < nst) {
            const bf16_t* an = ag + (size_t)(s + 1) * GBK;
            const bf16_t* bn = bg + (size_t)(s + 1) * GBK;
            ra0 = *(const bfx8*)an; ra1 = *(const bfx8*)(an + 8);
            rb0 = *(const bfx8*)bn; rb1 = *(const bfx8*)(bn + 8);
            if (s + 2 < nst) {
                __builtin_prefetch(ag + (size_t)(s + 2) * GBK, 0, 1);
                __builtin_prefetch(bg + (size_t)(s + 2) * GBK, 0, 1);
            }
        }
#endif

        // A fragment: lane l16 = row; 16 contiguous K at lh*16
        bfx16 af[4];
        #pragma unroll
        for (int m = 0; m < 4; ++m) {
            const bf16_t* p = &As[cur][(wm * 64 + m * 16 + l16) * GLD];
            ABfrag f;
            f.h[0] = *(const bfx8*)(p + lh * 16);
            f.h[1] = *(const bfx8*)(p + lh * 16 + 8);
            af[m] = f.v;
        }
        // B fragment: lane l16 = output column; 16 contiguous K at lh*16
        bfx16 bfr[2];
        #pragma unroll
        for (int n = 0; n < 2; ++n) {
            const bf16_t* p = &Bs[cur][(wn * 32 + n * 16 + l16) * GLD + lh * 16];
            ABfrag f;
            f.h[0] = *(const bfx8*)p;
            f.h[1] = *(const bfx8*)(p + 8);
            bfr[n] = f.v;
        }
        #pragma unroll
        for (int m = 0; m < 4; ++m)
            #pragma unroll
            for (int n = 0; n < 2; ++n)
                acc[m][n] = wmma_bf16(af[m], bfr[n], acc[m][n]);

        if (s + 1 < nst) {
#if defined(GEMM_USE_TDM)
            if (issuer) __builtin_amdgcn_s_wait_tensorcnt(0);
            __syncthreads();
#else
            __syncthreads();
            const int nxt = cur ^ 1;
            bf16_t* da = &As[nxt][lrow * GLD + lcol];
            bf16_t* db = &Bs[nxt][lrow * GLD + lcol];
            *(bfx8*)da = ra0; *(bfx8*)(da + 8) = ra1;
            *(bfx8*)db = rb0; *(bfx8*)(db + 8) = rb1;
            __syncthreads();
#endif
        }
    }

    // epilogue: C layout = row (lh*8+r), col l16
    #pragma unroll
    for (int m = 0; m < 4; ++m) {
        const int row0 = bm + wm * 64 + m * 16 + lh * 8;
        #pragma unroll
        for (int n = 0; n < 2; ++n) {
            const int col = bnb + wn * 32 + n * 16 + l16;
            #pragma unroll
            for (int r = 0; r < 8; ++r) {
                const size_t idx = (size_t)(row0 + r) * N + col;
                const float v = acc[m][n][r];
                if (EPI == EPI_BF16) {
                    ((bf16_t*)Cp)[idx] = to_bf(v);
                } else if (EPI == EPI_F32) {
                    ((float*)Cp)[idx] = v;
                } else if (EPI == EPI_RES) {
                    ((float*)Cp)[idx] += v;           // residual accumulate
                } else {
                    const float sv = v / (1.0f + __expf(-v));   // SiLU
                    ((bf16_t*)Cp)[idx] = to_bf(sv);
                }
            }
        }
    }
}

// ---------------------------------------------------------------------------
// Flash attention: block = (64 q-rows, 1 head, 1 batch), 128 threads = 4 waves.
// Each wave owns 16 q-rows: QK^T (4 WMMA k-steps over HEAD_DIM=128), online
// softmax with cross-lane reductions, P transposed via per-wave LDS, P*V WMMA.
// ---------------------------------------------------------------------------
#define AQT 64
#define AKT 32
#define KS_LD (HEAD_DIM + 8)  // 136
#define VT_LD (AKT + 8)       // 40
#define PS_LD (AKT + 8)       // 40

__global__ __launch_bounds__(128)
void attention_kernel(const bf16_t* __restrict__ Qb, const bf16_t* __restrict__ Kb,
                      const bf16_t* __restrict__ Vb, bf16_t* __restrict__ Ob) {
    __shared__ bf16_t Ks[AKT * KS_LD];        // K tile, row-major [key][hd]
    __shared__ bf16_t Vt[HEAD_DIM * VT_LD];   // V tile transposed [hd][key]
    __shared__ bf16_t Ps[4 * 16 * PS_LD];     // per-wave P scratch [qrow][key]

    const int tid  = threadIdx.x;
    const int wave = tid >> 5, lane = tid & 31;
    const int lh   = lane >> 4, l16 = lane & 15;
    const int qblk = blockIdx.x, head = blockIdx.y, batch = blockIdx.z;

    const size_t base = (size_t)batch * SEQ * D_MODEL + (size_t)head * HEAD_DIM;
    const bf16_t* Q = Qb + base;
    const bf16_t* K = Kb + base;
    const bf16_t* V = Vb + base;

    const int q0 = qblk * AQT + wave * 16;

    // Q fragments for this wave's 16 rows, 4 k-steps of 32 over HEAD_DIM
    bfx16 aq[4];
    {
        const bf16_t* qrow = Q + (size_t)(q0 + l16) * D_MODEL;
        #pragma unroll
        for (int ks = 0; ks < 4; ++ks) {
            ABfrag f;
            f.h[0] = *(const bfx8*)(qrow + ks * 32 + lh * 16);
            f.h[1] = *(const bfx8*)(qrow + ks * 32 + lh * 16 + 8);
            aq[ks] = f.v;
        }
    }

    fx8 o[8] = {};
    float mrun[8], lrun[8];
    #pragma unroll
    for (int r = 0; r < 8; ++r) { mrun[r] = -3.0e38f; lrun[r] = 0.0f; }

    const int nkv = (qblk + 1) * (AQT / AKT);   // causal: tiles 0..nkv-1
    for (int kt = 0; kt < nkv; ++kt) {
        const int kk = kt * AKT;

        // stage K tile (32 x 128), coalesced
        #pragma unroll
        for (int i = 0; i < 2; ++i) {
            const int seg = tid + i * 128;
            const int kr = seg >> 3, cs = (seg & 7) * 16;
            const bf16_t* src = K + (size_t)(kk + kr) * D_MODEL + cs;
            bf16_t* dst = Ks + kr * KS_LD + cs;
            *(bfx8*)dst       = *(const bfx8*)src;
            *(bfx8*)(dst + 8) = *(const bfx8*)(src + 8);
        }
        // stage V tile transposed: coalesced global, scattered ds stores
        {
            const int kr = tid >> 2;
            const int d0 = (tid & 3) * 32;
            const bf16_t* src = V + (size_t)(kk + kr) * D_MODEL + d0;
            bfx8 vv[4];
            #pragma unroll
            for (int j = 0; j < 4; ++j) vv[j] = ((const bfx8*)src)[j];
            #pragma unroll
            for (int j = 0; j < 4; ++j)
                #pragma unroll
                for (int e = 0; e < 8; ++e)
                    Vt[(d0 + j * 8 + e) * VT_LD + kr] = vv[j][e];
        }
        __syncthreads();

        // S = Q K^T : two 16-col fragments over the 32 keys
        fx8 s0 = {}, s1 = {};
        #pragma unroll
        for (int ks = 0; ks < 4; ++ks) {
            const bf16_t* kb0 = Ks + l16 * KS_LD + ks * 32 + lh * 16;
            const bf16_t* kb1 = Ks + (16 + l16) * KS_LD + ks * 32 + lh * 16;
            ABfrag b0, b1;
            b0.h[0] = *(const bfx8*)kb0; b0.h[1] = *(const bfx8*)(kb0 + 8);
            b1.h[0] = *(const bfx8*)kb1; b1.h[1] = *(const bfx8*)(kb1 + 8);
            s0 = wmma_bf16(aq[ks], b0.v, s0);
            s1 = wmma_bf16(aq[ks], b1.v, s1);
        }

        // online softmax (xor masks 1..8 stay within the 16-lane half)
        const float scl = 0.08838834764831845f;  // HEAD_DIM^-0.5
        float p0[8], p1[8], alpha[8];
        #pragma unroll
        for (int r = 0; r < 8; ++r) {
            const int qrow = qblk * AQT + wave * 16 + lh * 8 + r;
            float v0 = s0[r] * scl, v1 = s1[r] * scl;
            if (kk + l16 > qrow)      v0 = -3.0e38f;
            if (kk + 16 + l16 > qrow) v1 = -3.0e38f;
            float t = fmaxf(v0, v1);
            #pragma unroll
            for (int m = 8; m >= 1; m >>= 1) t = fmaxf(t, __shfl_xor(t, m, 32));
            const float mnew = fmaxf(mrun[r], t);
            alpha[r] = __expf(mrun[r] - mnew);
            mrun[r] = mnew;
            p0[r] = __expf(v0 - mnew);
            p1[r] = __expf(v1 - mnew);
            float rs = p0[r] + p1[r];
            #pragma unroll
            for (int m = 8; m >= 1; m >>= 1) rs += __shfl_xor(rs, m, 32);
            lrun[r] = lrun[r] * alpha[r] + rs;
        }
        #pragma unroll
        for (int nf = 0; nf < 8; ++nf)
            #pragma unroll
            for (int r = 0; r < 8; ++r) o[nf][r] = o[nf][r] * alpha[r];

        // transpose P (C layout -> A layout) through per-wave LDS scratch
        bf16_t* ps = Ps + wave * 16 * PS_LD;
        #pragma unroll
        for (int r = 0; r < 8; ++r) {
            ps[(lh * 8 + r) * PS_LD + l16]      = to_bf(p0[r]);
            ps[(lh * 8 + r) * PS_LD + 16 + l16] = to_bf(p1[r]);
        }
        ABfrag pf;
        pf.h[0] = *(const bfx8*)(ps + l16 * PS_LD + lh * 16);
        pf.h[1] = *(const bfx8*)(ps + l16 * PS_LD + lh * 16 + 8);

        // O += P * V  (8 fragments across HEAD_DIM)
        #pragma unroll
        for (int nf = 0; nf < 8; ++nf) {
            const bf16_t* vp = Vt + (nf * 16 + l16) * VT_LD + lh * 16;
            ABfrag vf;
            vf.h[0] = *(const bfx8*)vp; vf.h[1] = *(const bfx8*)(vp + 8);
            o[nf] = wmma_bf16(pf.v, vf.v, o[nf]);
        }
        __syncthreads();
    }

    // normalize + store
    bf16_t* Orow = Ob + base;
    #pragma unroll
    for (int r = 0; r < 8; ++r) {
        const int qrow = qblk * AQT + wave * 16 + lh * 8 + r;
        const float inv = 1.0f / lrun[r];
        #pragma unroll
        for (int nf = 0; nf < 8; ++nf)
            Orow[(size_t)qrow * D_MODEL + nf * 16 + l16] = to_bf(o[nf][r] * inv);
    }
}

// ---------------------------------------------------------------------------
// Host-side orchestration
// ---------------------------------------------------------------------------
extern "C" void kernel_launch(void* const* d_in, const int* in_sizes, int n_in,
                              void* d_out, int out_size, void* d_ws, size_t ws_size,
                              hipStream_t stream) {
    (void)in_sizes; (void)n_in; (void)out_size; (void)ws_size;

    const int*   tokens   = (const int*)  d_in[0];
    const float* tok_emb  = (const float*)d_in[1];
    const float* wq       = (const float*)d_in[2];
    const float* wk       = (const float*)d_in[3];
    const float* wv       = (const float*)d_in[4];
    const float* wo       = (const float*)d_in[5];
    const float* attn_nw  = (const float*)d_in[6];
    const float* ffn_nw   = (const float*)d_in[7];
    const float* final_nw = (const float*)d_in[8];
    const float* w1       = (const float*)d_in[9];
    const float* w2       = (const float*)d_in[10];
    const float* out_proj = (const float*)d_in[11];
    float* logits = (float*)d_out;

    char* ws = (char*)d_ws;
    size_t off = 0;
    auto alloc = [&](size_t bytes) -> char* {
        char* p = ws + off;
        off += (bytes + 255) & ~(size_t)255;
        return p;
    };

    float*  x   = (float*) alloc((size_t)NTOK * D_MODEL * 4);   // residual stream
    bf16_t* xn  = (bf16_t*)alloc((size_t)NTOK * D_MODEL * 2);   // normed acts
    bf16_t* qb  = (bf16_t*)alloc((size_t)NTOK * D_MODEL * 2);
    bf16_t* kb  = (bf16_t*)alloc((size_t)NTOK * D_MODEL * 2);
    bf16_t* vb  = (bf16_t*)alloc((size_t)NTOK * D_MODEL * 2);
    bf16_t* ao  = (bf16_t*)alloc((size_t)NTOK * D_MODEL * 2);   // attn output
    bf16_t* hb  = (bf16_t*)alloc((size_t)NTOK * D_FF * 2);      // FFN hidden
    bf16_t* wqb = (bf16_t*)alloc((size_t)D_MODEL * D_MODEL * 2);
    bf16_t* wkb = (bf16_t*)alloc((size_t)D_MODEL * D_MODEL * 2);
    bf16_t* wvb = (bf16_t*)alloc((size_t)D_MODEL * D_MODEL * 2);
    bf16_t* wob = (bf16_t*)alloc((size_t)D_MODEL * D_MODEL * 2);
    bf16_t* w1b = (bf16_t*)alloc((size_t)D_FF * D_MODEL * 2);
    bf16_t* w2b = (bf16_t*)alloc((size_t)D_MODEL * D_FF * 2);
    bf16_t* opb = (bf16_t*)alloc((size_t)VOCAB * D_MODEL * 2);

    auto cvt = [&](const float* src, bf16_t* dst, size_t n) {
        cvt_f32_to_bf16<<<(unsigned)(n / 1024), 256, 0, stream>>>(src, dst, (int)n);
    };
    cvt(wq, wqb, (size_t)D_MODEL * D_MODEL);
    cvt(wk, wkb, (size_t)D_MODEL * D_MODEL);
    cvt(wv, wvb, (size_t)D_MODEL * D_MODEL);
    cvt(wo, wob, (size_t)D_MODEL * D_MODEL);
    cvt(w1, w1b, (size_t)D_FF * D_MODEL);
    cvt(w2, w2b, (size_t)D_MODEL * D_FF);
    cvt(out_proj, opb, (size_t)VOCAB * D_MODEL);

    embed_kernel<<<NTOK, 256, 0, stream>>>(tokens, tok_emb, x);

    // --- attention block ---
    rmsnorm_kernel<<<NTOK, 256, 0, stream>>>(x, attn_nw, xn);
    gemm_bf16_wmma<EPI_BF16><<<dim3(D_MODEL / GBN, NTOK / GBM), 256, 0, stream>>>(
        xn, wqb, qb, NTOK, D_MODEL, D_MODEL);
    gemm_bf16_wmma<EPI_BF16><<<dim3(D_MODEL / GBN, NTOK / GBM), 256, 0, stream>>>(
        xn, wkb, kb, NTOK, D_MODEL, D_MODEL);
    gemm_bf16_wmma<EPI_BF16><<<dim3(D_MODEL / GBN, NTOK / GBM), 256, 0, stream>>>(
        xn, wvb, vb, NTOK, D_MODEL, D_MODEL);
    attention_kernel<<<dim3(SEQ / AQT, N_HEADS, BATCH), 128, 0, stream>>>(qb, kb, vb, ao);
    gemm_bf16_wmma<EPI_RES><<<dim3(D_MODEL / GBN, NTOK / GBM), 256, 0, stream>>>(
        ao, wob, x, NTOK, D_MODEL, D_MODEL);

    // --- FFN block ---
    rmsnorm_kernel<<<NTOK, 256, 0, stream>>>(x, ffn_nw, xn);
    gemm_bf16_wmma<EPI_SILU><<<dim3(D_FF / GBN, NTOK / GBM), 256, 0, stream>>>(
        xn, w1b, hb, NTOK, D_FF, D_MODEL);
    gemm_bf16_wmma<EPI_RES><<<dim3(D_MODEL / GBN, NTOK / GBM), 256, 0, stream>>>(
        hb, w2b, x, NTOK, D_MODEL, D_FF);

    // --- LM head ---
    rmsnorm_kernel<<<NTOK, 256, 0, stream>>>(x, final_nw, xn);
    gemm_bf16_wmma<EPI_F32><<<dim3(VOCAB / GBN, NTOK / GBM), 256, 0, stream>>>(
        xn, opb, logits, NTOK, VOCAB, D_MODEL);
}
